// GraphAttentionLayer_9397388444030
// MI455X (gfx1250) — matrix operand order repaired and checked by reading, not compiled, not run on hardware
//
#include <hip/hip_runtime.h>
#include <hip/hip_bf16.h>

typedef __attribute__((ext_vector_type(16))) _Float16 v16h;
typedef __attribute__((ext_vector_type(8)))  float    v8f;
typedef __attribute__((ext_vector_type(2)))  float    v2f;

#define NN 8192
#define F_IN 128
#define F_OUT 64
#define ALPHA 0.2f
#define MASK_VAL -9000000000000000.0f

// ---------------------------------------------------------------------------
// Kernel 1: wh = h_nodes @ weight  (fp32 WMMA 16x16x4), also emit whT in f16,
// transposed [64][8192], so the attention kernel's B tiles are contiguous.
// Block = 128 threads (4 waves), one wave per 16-col N tile, 16 rows/block.
// ---------------------------------------------------------------------------
__global__ __launch_bounds__(128) void wh_kernel(
    const float* __restrict__ h, const float* __restrict__ w,
    float* __restrict__ wh, _Float16* __restrict__ whT) {
  const int wave = threadIdx.x >> 5;     // N tile 0..3
  const int lane = threadIdx.x & 31;
  const int hi   = lane >> 4;            // lane half
  const int m    = lane & 15;            // A row (lanes 0-15 M=0..15)
  const int rowbase = blockIdx.x * 16;
  const int n = wave * 16 + m;           // global output column (B: N = lane&15)

  v8f c = {};
  for (int k = 0; k < F_IN; k += 4) {
    // A 16x4 f32: v0 = K (0|2 per lane half), v1 = K+1|K+3  -> contiguous pair
    v2f a = *(const v2f*)(h + (size_t)(rowbase + m) * F_IN + k + 2 * hi);
    // B 4x16 f32: v0 = row K (0|2), v1 = row K+1|K+3, col = lane&15
    v2f b;
    b.x = w[(size_t)(k + 2 * hi) * F_OUT + n];
    b.y = w[(size_t)(k + 2 * hi + 1) * F_OUT + n];
    c = __builtin_amdgcn_wmma_f32_16x16x4_f32(false, a, false, b,
                                              (short)0, c, false, false);
  }
#pragma unroll
  for (int r = 0; r < 8; ++r) {
    const int M = r + 8 * hi;            // C layout: M = vgpr + 8*(lane>=16)
    const float v = c[r];
    wh[(size_t)(rowbase + M) * F_OUT + n] = v;
    whT[(size_t)n * NN + rowbase + M] = (_Float16)v;
  }
}

// ---------------------------------------------------------------------------
// Kernel 2: src[i] = wh[i]·a1 + edge[i]*(w_edge·a3) ; dst[i] = wh[i]·a2
// ---------------------------------------------------------------------------
__global__ __launch_bounds__(256) void srcdst_kernel(
    const float* __restrict__ wh, const float* __restrict__ edge,
    const float* __restrict__ att, const float* __restrict__ w_edge,
    float* __restrict__ srcv, float* __restrict__ dstv) {
  const int i = blockIdx.x * 256 + threadIdx.x;
  const float* whrow = wh + (size_t)i * F_OUT;
  float s1 = 0.f, s2 = 0.f, sE = 0.f;
#pragma unroll 8
  for (int nc = 0; nc < F_OUT; ++nc) {
    const float v = whrow[nc];
    s1 += v * att[nc];
    s2 += v * att[F_OUT + nc];
    sE += w_edge[nc] * att[2 * F_OUT + nc];
  }
  srcv[i] = s1 + edge[i] * sE;
  dstv[i] = s2;
}

// ---------------------------------------------------------------------------
// Kernel 3: fused masked-softmax attention + P@wh + ELU.
// One block = 16 rows, 8 waves. Pass A: per-row online (max, sumexp).
// Pass B: per-wave column slab of 1024, f16 WMMA 16x16x32 accumulation.
// ---------------------------------------------------------------------------
__global__ __launch_bounds__(256) void gat_kernel(
    const int* __restrict__ adj, const float* __restrict__ srcv,
    const float* __restrict__ dstv, const _Float16* __restrict__ whT,
    float* __restrict__ out) {
  __shared__ float rowMax[16];
  __shared__ float rowSum[16];
  __shared__ float accLds[16 * F_OUT];

  const int tid = threadIdx.x;
  const int wave = tid >> 5;
  const int lane = tid & 31;
  const int hi = lane >> 4;
  const int rowbase = blockIdx.x * 16;

  for (int t = tid; t < 16 * F_OUT; t += 256) accLds[t] = 0.f;

  // ---- Pass A: per-row max and sum(exp) of masked leaky-relu logits ----
  for (int rr = 0; rr < 2; ++rr) {
    const int row = 2 * wave + rr;
    const float s_i = srcv[rowbase + row];
    const int* arow = adj + (size_t)(rowbase + row) * NN;
    float m = -INFINITY, s = 0.f;
    for (int k = 0; k < NN / 128; ++k) {
      const int j0 = k * 128 + lane * 4;
      const int4 a4 = *(const int4*)(arow + j0);
      const float4 d4 = *(const float4*)(dstv + j0);
      const float dv[4] = {d4.x, d4.y, d4.z, d4.w};
      const int av[4] = {a4.x, a4.y, a4.z, a4.w};
#pragma unroll
      for (int q = 0; q < 4; ++q) {
        const float raw = s_i + dv[q];
        float e = raw > 0.f ? raw : ALPHA * raw;
        e = (av[q] > 0) ? e : MASK_VAL;
        const float Mn = fmaxf(m, e);
        s = s * __expf(m - Mn) + __expf(e - Mn);
        m = Mn;
      }
    }
#pragma unroll
    for (int off = 16; off > 0; off >>= 1) {
      const float m2 = __shfl_xor(m, off, 32);
      const float s2 = __shfl_xor(s, off, 32);
      const float Mn = fmaxf(m, m2);
      s = s * __expf(m - Mn) + s2 * __expf(m2 - Mn);
      m = Mn;
    }
    if (lane == 0) { rowMax[row] = m; rowSum[row] = s; }
  }
  __syncthreads();

  // ---- Pass B: fused P (f16, A-matrix layout) @ wh (f16 B tiles from whT) --
  const int myrow = lane & 15;                 // A layout: M = lane&15
  const float mrow = rowMax[myrow];
  const float inv_s = 1.0f / rowSum[myrow];
  const float s_row = srcv[rowbase + myrow];
  const int* arow = adj + (size_t)(rowbase + myrow) * NN;

  v8f c0 = {}, c1 = {}, c2 = {}, c3 = {};
  const int colBeg = wave * (NN / 8);
  for (int chunk = colBeg; chunk < colBeg + NN / 8; chunk += 32) {
    // 16 contiguous K per lane half (K = 16*hi .. +15), matching B
    const int b1 = chunk + 16 * hi;            // K = 16*hi .. +7
    const int b2 = b1 + 8;                     // K = 16*hi+8 .. +15
    const int4 q0 = *(const int4*)(arow + b1);
    const int4 q1 = *(const int4*)(arow + b1 + 4);
    const int4 q2 = *(const int4*)(arow + b2);
    const int4 q3 = *(const int4*)(arow + b2 + 4);
    const float4 d0 = *(const float4*)(dstv + b1);
    const float4 d1 = *(const float4*)(dstv + b1 + 4);
    const float4 d2 = *(const float4*)(dstv + b2);
    const float4 d3 = *(const float4*)(dstv + b2 + 4);
    const float dd[16] = {d0.x, d0.y, d0.z, d0.w, d1.x, d1.y, d1.z, d1.w,
                          d2.x, d2.y, d2.z, d2.w, d3.x, d3.y, d3.z, d3.w};
    const int av[16] = {q0.x, q0.y, q0.z, q0.w, q1.x, q1.y, q1.z, q1.w,
                        q2.x, q2.y, q2.z, q2.w, q3.x, q3.y, q3.z, q3.w};
    v16h A;
#pragma unroll
    for (int q = 0; q < 16; ++q) {
      const float raw = s_row + dd[q];
      float e = raw > 0.f ? raw : ALPHA * raw;
      e = (av[q] > 0) ? e : MASK_VAL;
      A[q] = (_Float16)(__expf(e - mrow) * inv_s);   // 1/s folded per row
    }
    // B 32x16 f16 in this code's k order: lane = column N, 16 contiguous f16 along K (split
    // by lane half) -> direct 32B load from transposed whT.
    const size_t joff = (size_t)chunk + 16 * hi;
    const v16h B0 = *(const v16h*)(whT + (size_t)(0 * 16 + myrow) * NN + joff);
    const v16h B1 = *(const v16h*)(whT + (size_t)(1 * 16 + myrow) * NN + joff);
    const v16h B2 = *(const v16h*)(whT + (size_t)(2 * 16 + myrow) * NN + joff);
    const v16h B3 = *(const v16h*)(whT + (size_t)(3 * 16 + myrow) * NN + joff);
    c0 = __builtin_amdgcn_wmma_f32_16x16x32_f16(false, A, false, B0, (short)0, c0, false, false);
    c1 = __builtin_amdgcn_wmma_f32_16x16x32_f16(false, A, false, B1, (short)0, c1, false, false);
    c2 = __builtin_amdgcn_wmma_f32_16x16x32_f16(false, A, false, B2, (short)0, c2, false, false);
    c3 = __builtin_amdgcn_wmma_f32_16x16x32_f16(false, A, false, B3, (short)0, c3, false, false);
  }

  // ---- Cross-wave reduction in LDS (ds_add_f32) ----
#pragma unroll
  for (int r = 0; r < 8; ++r) {
    const int M = r + 8 * hi;
    atomicAdd(&accLds[M * F_OUT + 0 * 16 + myrow], c0[r]);
    atomicAdd(&accLds[M * F_OUT + 1 * 16 + myrow], c1[r]);
    atomicAdd(&accLds[M * F_OUT + 2 * 16 + myrow], c2[r]);
    atomicAdd(&accLds[M * F_OUT + 3 * 16 + myrow], c3[r]);
  }
  __syncthreads();

  // ---- ELU epilogue + store ----
  for (int t = tid; t < 16 * F_OUT; t += 256) {
    const float v = accLds[t];
    out[(size_t)rowbase * F_OUT + t] = v > 0.f ? v : (__expf(v) - 1.0f);
  }
}

extern "C" void kernel_launch(void* const* d_in, const int* in_sizes, int n_in,
                              void* d_out, int out_size, void* d_ws, size_t ws_size,
                              hipStream_t stream) {
  const float* h_nodes = (const float*)d_in[0];
  const float* edge    = (const float*)d_in[1];
  const int*   adj     = (const int*)d_in[2];
  const float* weight  = (const float*)d_in[3];
  const float* att     = (const float*)d_in[4];
  const float* w_edge  = (const float*)d_in[5];
  float* out = (float*)d_out;

  char* ws = (char*)d_ws;
  float*    wh   = (float*)ws;                                  // 8192*64*4 = 2 MB
  _Float16* whT  = (_Float16*)(ws + (size_t)2097152);           // 64*8192*2 = 1 MB
  float*    srcv = (float*)(ws + (size_t)3145728);              // 32 KB
  float*    dstv = (float*)(ws + (size_t)3178496);              // 32 KB

  wh_kernel<<<NN / 16, 128, 0, stream>>>(h_nodes, weight, wh, whT);
  srcdst_kernel<<<NN / 256, 256, 0, stream>>>(wh, edge, att, w_edge, srcv, dstv);
  gat_kernel<<<NN / 16, 256, 0, stream>>>(adj, srcv, dstv, whT, out);
}
